// Model_90744069030634
// MI455X (gfx1250) — matrix operand order repaired and checked by reading, not compiled, not run on hardware
//
#include <hip/hip_runtime.h>
#include <stdint.h>
#include <stddef.h>

// ---------------------------------------------------------------------------
// Types for CDNA5 WMMA (wave32): D = A(16x32 bf16) x B(32x16 bf16) + C(16x16 f32)
// ---------------------------------------------------------------------------
typedef __attribute__((ext_vector_type(16))) __bf16 v16bf;
typedef __attribute__((ext_vector_type(8)))  float  v8f;

union Frag32B { uint4 q[2]; v16bf v; };

// A-matrix 16x32 bf16 per-lane fragment (contiguous k order used here, same as load_b_frag):
// lane<16  : row m=lane,    K = [k0..k0+15]
// lane>=16 : row m=lane-16, K = [k0+16..k0+31]
__device__ __forceinline__ v16bf load_a_frag(const __bf16* rowPtr, int k0, int khalf) {
  Frag32B f;
  f.q[0] = *(const uint4*)(rowPtr + k0 + 16 * khalf);
  f.q[1] = *(const uint4*)(rowPtr + k0 + 8 + 16 * khalf);
  return f.v;
}

// B-matrix 32x16 bf16 per-lane fragment: lane owns column n=lane&15,
// lanes 0-15 hold K=[k0..k0+15], lanes 16-31 hold K=[k0+16..k0+31].
// Weights are pre-transposed (column-major: Wt[n*K + k]) so this is contiguous.
__device__ __forceinline__ v16bf load_b_frag(const __bf16* colPtr, int k0, int khalf) {
  Frag32B f;
  const __bf16* p = colPtr + k0 + 16 * khalf;
  f.q[0] = *(const uint4*)(p);
  f.q[1] = *(const uint4*)(p + 8);
  return f.v;
}

// ---------------------------------------------------------------------------
// Utility kernels
// ---------------------------------------------------------------------------
__global__ void fill_zero_kernel(float* p, long long n) {
  long long i = (long long)blockIdx.x * blockDim.x + threadIdx.x;
  long long s = (long long)gridDim.x * blockDim.x;
  for (; i < n; i += s) p[i] = 0.0f;
}

__global__ void cvt_f32_bf16_kernel(const float* __restrict__ in, __bf16* __restrict__ out, long long n) {
  long long i = (long long)blockIdx.x * blockDim.x + threadIdx.x;
  long long s = (long long)gridDim.x * blockDim.x;
  for (; i < n; i += s) out[i] = (__bf16)in[i];
}

// Wt[n*K + k] = (bf16) W[k*N + n]   (W is row-major [K,N] as in the reference)
__global__ void transpose_bf16_kernel(const float* __restrict__ W, __bf16* __restrict__ Wt, int K, int N) {
  int i = blockIdx.x * blockDim.x + threadIdx.x;
  int total = K * N;
  if (i < total) {
    int k = i / N, n = i % N;
    Wt[(size_t)n * K + k] = (__bf16)W[i];
  }
}

// out[row,f] = agg[row,f] / max(cnt[row],1)  -> bf16
__global__ void mean_cvt_kernel(const float* __restrict__ agg, const float* __restrict__ cnt,
                                __bf16* __restrict__ out, long long rows) {
  long long n = rows * 256;
  long long i = (long long)blockIdx.x * blockDim.x + threadIdx.x;
  long long s = (long long)gridDim.x * blockDim.x;
  for (; i < n; i += s) {
    long long row = i >> 8;
    float c = cnt[row];
    out[i] = (__bf16)(agg[i] / fmaxf(c, 1.0f));
  }
}

// ---------------------------------------------------------------------------
// Edge scatter: per edge e, agg_s[dst] += xd[src], agg_d[src] += xs[dst].
// 256 threads = 256 features -> coalesced f32 atomics (destinations are
// L2-resident: 82 MB << 192 MB L2). Counts only computed when doCount!=0.
// ---------------------------------------------------------------------------
__global__ void scatter_kernel(const __bf16* __restrict__ xd, const __bf16* __restrict__ xs,
                               const int* __restrict__ src, const int* __restrict__ dst,
                               float* __restrict__ agg_s, float* __restrict__ agg_d,
                               float* __restrict__ cnt_s, float* __restrict__ cnt_d,
                               int E, int doCount) {
  const int t = threadIdx.x;  // 0..255
  for (int e = blockIdx.x; e < E; e += gridDim.x) {
    int s = src[e];
    int d = dst[e];
    float vd = (float)xd[(size_t)s * 256 + t];
    atomicAdd(agg_s + (size_t)d * 256 + t, vd);
    float vs = (float)xs[(size_t)d * 256 + t];
    atomicAdd(agg_d + (size_t)s * 256 + t, vs);
    if (doCount) {
      if (t == 0) atomicAdd(cnt_s + d, 1.0f);
      else if (t == 1) atomicAdd(cnt_d + s, 1.0f);
    }
  }
}

// ---------------------------------------------------------------------------
// Fused SAGE transform: Out = act( Aagg@Wl + bias + Xself@Wr )  (K=N=256)
// 256 threads = 8 waves; each wave computes a 16x256 output strip with
// 16 f32 accumulators (v8f acc[16]) and v_wmma_f32_16x16x32_bf16.
// __launch_bounds__(256,1) keeps the 128-VGPR accumulator array resident.
// Fragments are software-pipelined one step ahead so a B-load is always in
// flight behind each WMMA (wait becomes loadcnt<=1 instead of 0).
// ---------------------------------------------------------------------------
__device__ __forceinline__ void gemm_pass256(const __bf16* __restrict__ rowPtr,
                                             const __bf16* __restrict__ Wt,
                                             int khalf, int nsub, v8f acc[16]) {
  const __bf16* colBase = Wt + (size_t)nsub * 256;  // column nsub; tile stride 16*256 elems
  v16bf a_cur = load_a_frag(rowPtr, 0, khalf);
  v16bf b_cur = load_b_frag(colBase, 0, khalf);
  for (int k0 = 0; k0 < 256; k0 += 32) {
    v16bf a_nxt = a_cur;
    if (k0 < 224) a_nxt = load_a_frag(rowPtr, k0 + 32, khalf);
#pragma unroll
    for (int nt = 0; nt < 16; ++nt) {
      v16bf b_nxt = b_cur;
      if (nt < 15) {
        b_nxt = load_b_frag(colBase + (size_t)(nt + 1) * (16 * 256), k0, khalf);
      } else if (k0 < 224) {
        b_nxt = load_b_frag(colBase, k0 + 32, khalf);
      }
      acc[nt] = __builtin_amdgcn_wmma_f32_16x16x32_bf16(false, a_cur, false, b_cur,
                                                        (short)0, acc[nt], false, false);
      b_cur = b_nxt;
    }
    a_cur = a_nxt;
  }
}

__global__ void __launch_bounds__(256, 1)
sage_wmma_kernel(const __bf16* __restrict__ Aagg, const __bf16* __restrict__ Xself,
                 const __bf16* __restrict__ Wl_t, const __bf16* __restrict__ Wr_t,
                 const float* __restrict__ bias, __bf16* __restrict__ Out,
                 int M, int doRelu) {
  const int lane  = threadIdx.x & 31;
  const int wave  = threadIdx.x >> 5;
  const int khalf = lane >> 4;
  const int nsub  = lane & 15;
  const int rowBase = blockIdx.x * 128 + wave * 16;

  int arow = rowBase + nsub;
  if (arow > M - 1) arow = M - 1;  // clamp loads; WMMA needs EXEC all-ones

  v8f acc[16];
#pragma unroll
  for (int nt = 0; nt < 16; ++nt) {
    float bv = bias[nt * 16 + nsub];
#pragma unroll
    for (int i = 0; i < 8; ++i) acc[nt][i] = bv;
  }

  gemm_pass256(Aagg  + (size_t)arow * 256, Wl_t, khalf, nsub, acc);
  gemm_pass256(Xself + (size_t)arow * 256, Wr_t, khalf, nsub, acc);

#pragma unroll
  for (int nt = 0; nt < 16; ++nt) {
#pragma unroll
    for (int r = 0; r < 8; ++r) {
      int row = rowBase + r + 8 * khalf;  // D layout: lanes<16 -> m=r, lanes>=16 -> m=r+8
      if (row < M) {
        float v = acc[nt][r];
        if (doRelu) v = fmaxf(v, 0.0f);
        Out[(size_t)row * 256 + nt * 16 + nsub] = (__bf16)v;
      }
    }
  }
}

// ---------------------------------------------------------------------------
// Fused decoder: out = relu(concat(zd[lrow], zs[lcol]) @ W1 + b1) @ W2 + b2.
// 128 threads = 4 waves, 64 label rows/block. Gather rows into 32 KB LDS in two
// K-phases (drug half, disease half); WMMA K-loop reads A from LDS; the [256,1]
// second matmul is folded into per-lane dots + shfl_xor reduction (h never
// touches memory). Same 1-deep B-fragment pipelining as the SAGE kernel.
// ---------------------------------------------------------------------------
__global__ void __launch_bounds__(128, 1)
decoder_kernel(const __bf16* __restrict__ zd, const __bf16* __restrict__ zs,
               const int* __restrict__ lrow, const int* __restrict__ lcol,
               const __bf16* __restrict__ W1t, const float* __restrict__ b1,
               const float* __restrict__ W2, const float* __restrict__ b2,
               float* __restrict__ out, int EL) {
  __shared__ __align__(16) __bf16 sA[64 * 256];  // 32 KB

  const int t     = threadIdx.x;  // 0..127
  const int lane  = t & 31;
  const int wave  = t >> 5;       // 0..3
  const int khalf = lane >> 4;
  const int nsub  = lane & 15;
  const int blockBase = blockIdx.x * 64;

  v8f acc[16];
#pragma unroll
  for (int nt = 0; nt < 16; ++nt) {
    float bv = b1[nt * 16 + nsub];
#pragma unroll
    for (int i = 0; i < 8; ++i) acc[nt][i] = bv;
  }

  const int grow = t >> 1;   // local row for staging (2 threads/row)
  const int part = t & 1;    // which 128-feature half this thread stages
  int gidx = blockBase + grow;
  if (gidx > EL - 1) gidx = EL - 1;

  for (int phase = 0; phase < 2; ++phase) {
    const __bf16* srcRow = phase ? (zs + (size_t)lcol[gidx] * 256)
                                 : (zd + (size_t)lrow[gidx] * 256);
    const uint4* sg = (const uint4*)(srcRow + part * 128);
    uint4* dl = (uint4*)(sA + grow * 256 + part * 128);
#pragma unroll
    for (int i = 0; i < 16; ++i) dl[i] = sg[i];
    __syncthreads();

    const __bf16* aRow = sA + (size_t)(wave * 16 + nsub) * 256;
    const __bf16* colBase = W1t + (size_t)nsub * 512 + phase * 256;  // tile stride 16*512
    v16bf a_cur = load_a_frag(aRow, 0, khalf);
    v16bf b_cur = load_b_frag(colBase, 0, khalf);
    for (int k0 = 0; k0 < 256; k0 += 32) {
      v16bf a_nxt = a_cur;
      if (k0 < 224) a_nxt = load_a_frag(aRow, k0 + 32, khalf);
#pragma unroll
      for (int nt = 0; nt < 16; ++nt) {
        v16bf b_nxt = b_cur;
        if (nt < 15) {
          b_nxt = load_b_frag(colBase + (size_t)(nt + 1) * (16 * 512), k0, khalf);
        } else if (k0 < 224) {
          b_nxt = load_b_frag(colBase, k0 + 32, khalf);
        }
        acc[nt] = __builtin_amdgcn_wmma_f32_16x16x32_bf16(false, a_cur, false, b_cur,
                                                          (short)0, acc[nt], false, false);
        b_cur = b_nxt;
      }
      a_cur = a_nxt;
    }
    __syncthreads();  // protect sA before next phase overwrites it
  }

  // second matmul: per-lane dot over owned columns, then 16-lane reduction
  float partial[8];
#pragma unroll
  for (int r = 0; r < 8; ++r) partial[r] = 0.0f;
#pragma unroll
  for (int nt = 0; nt < 16; ++nt) {
    float w2 = W2[nt * 16 + nsub];
#pragma unroll
    for (int r = 0; r < 8; ++r) partial[r] += fmaxf(acc[nt][r], 0.0f) * w2;
  }
#pragma unroll
  for (int m = 1; m < 16; m <<= 1) {
#pragma unroll
    for (int r = 0; r < 8; ++r) partial[r] += __shfl_xor(partial[r], m, 32);
  }
  if (nsub == 0) {
    float bb = b2[0];
#pragma unroll
    for (int r = 0; r < 8; ++r) {
      int row = blockBase + wave * 16 + r + 8 * khalf;
      if (row < EL) out[row] = partial[r] + bb;
    }
  }
}

// ---------------------------------------------------------------------------
// Host-side orchestration
// ---------------------------------------------------------------------------
extern "C" void kernel_launch(void* const* d_in, const int* in_sizes, int n_in,
                              void* d_out, int out_size, void* d_ws, size_t ws_size,
                              hipStream_t stream) {
  const float* x_drug = (const float*)d_in[0];
  const float* x_dis  = (const float*)d_in[1];
  const int*   esrc   = (const int*)d_in[2];
  const int*   edst   = (const int*)d_in[3];
  const int*   lrow   = (const int*)d_in[4];
  const int*   lcol   = (const int*)d_in[5];
  const float* Wl1_ds = (const float*)d_in[6];
  const float* bl1_ds = (const float*)d_in[7];
  const float* Wr1_ds = (const float*)d_in[8];
  const float* Wl1_sd = (const float*)d_in[9];
  const float* bl1_sd = (const float*)d_in[10];
  const float* Wr1_sd = (const float*)d_in[11];
  const float* Wl2_ds = (const float*)d_in[12];
  const float* bl2_ds = (const float*)d_in[13];
  const float* Wr2_ds = (const float*)d_in[14];
  const float* Wl2_sd = (const float*)d_in[15];
  const float* bl2_sd = (const float*)d_in[16];
  const float* Wr2_sd = (const float*)d_in[17];
  const float* Wdec1  = (const float*)d_in[18];
  const float* bdec1  = (const float*)d_in[19];
  const float* Wdec2  = (const float*)d_in[20];
  const float* bdec2  = (const float*)d_in[21];
  float* out = (float*)d_out;

  const int NDn = in_sizes[0] / 256;   // 50000
  const int NSn = in_sizes[1] / 256;   // 30000
  const int E   = in_sizes[2];         // 1,000,000
  const int EL  = in_sizes[4];         // 200,000

  // ---- workspace carving (256B aligned) ----
  char* ws = (char*)d_ws;
  size_t off = 0;
  auto carve = [&](size_t bytes) -> char* {
    char* p = ws + off;
    off += (bytes + 255) & ~(size_t)255;
    return p;
  };
  __bf16* xd_bf   = (__bf16*)carve((size_t)NDn * 256 * 2);
  __bf16* xs_bf   = (__bf16*)carve((size_t)NSn * 256 * 2);
  __bf16* wl1ds_t = (__bf16*)carve(256 * 256 * 2);
  __bf16* wr1ds_t = (__bf16*)carve(256 * 256 * 2);
  __bf16* wl1sd_t = (__bf16*)carve(256 * 256 * 2);
  __bf16* wr1sd_t = (__bf16*)carve(256 * 256 * 2);
  __bf16* wl2ds_t = (__bf16*)carve(256 * 256 * 2);
  __bf16* wr2ds_t = (__bf16*)carve(256 * 256 * 2);
  __bf16* wl2sd_t = (__bf16*)carve(256 * 256 * 2);
  __bf16* wr2sd_t = (__bf16*)carve(256 * 256 * 2);
  __bf16* wdec1_t = (__bf16*)carve(512 * 256 * 2);
  float*  agg_s   = (float*)carve((size_t)NSn * 256 * 4);
  float*  agg_d   = (float*)carve((size_t)NDn * 256 * 4);
  float*  cnt_s   = (float*)carve((size_t)NSn * 4);
  float*  cnt_d   = (float*)carve((size_t)NDn * 4);
  __bf16* aggs_bf = (__bf16*)carve((size_t)NSn * 256 * 2);
  __bf16* aggd_bf = (__bf16*)carve((size_t)NDn * 256 * 2);
  __bf16* z1d     = (__bf16*)carve((size_t)NDn * 256 * 2);
  __bf16* z1s     = (__bf16*)carve((size_t)NSn * 256 * 2);
  __bf16* z2d     = xd_bf;  // x bf16 buffers are dead after layer 1 -> reuse
  __bf16* z2s     = xs_bf;
  (void)ws_size; (void)n_in; (void)out_size;

  const long long nD = (long long)NDn * 256;
  const long long nS = (long long)NSn * 256;

  // ---- prep: bf16 conversions + weight transposes ----
  cvt_f32_bf16_kernel<<<4096, 256, 0, stream>>>(x_drug, xd_bf, nD);
  cvt_f32_bf16_kernel<<<4096, 256, 0, stream>>>(x_dis,  xs_bf, nS);
  transpose_bf16_kernel<<<256, 256, 0, stream>>>(Wl1_ds, wl1ds_t, 256, 256);
  transpose_bf16_kernel<<<256, 256, 0, stream>>>(Wr1_ds, wr1ds_t, 256, 256);
  transpose_bf16_kernel<<<256, 256, 0, stream>>>(Wl1_sd, wl1sd_t, 256, 256);
  transpose_bf16_kernel<<<256, 256, 0, stream>>>(Wr1_sd, wr1sd_t, 256, 256);
  transpose_bf16_kernel<<<256, 256, 0, stream>>>(Wl2_ds, wl2ds_t, 256, 256);
  transpose_bf16_kernel<<<256, 256, 0, stream>>>(Wr2_ds, wr2ds_t, 256, 256);
  transpose_bf16_kernel<<<256, 256, 0, stream>>>(Wl2_sd, wl2sd_t, 256, 256);
  transpose_bf16_kernel<<<256, 256, 0, stream>>>(Wr2_sd, wr2sd_t, 256, 256);
  transpose_bf16_kernel<<<512, 256, 0, stream>>>(Wdec1,  wdec1_t, 512, 256);

  // ---- layer 1 ----
  fill_zero_kernel<<<2048, 256, 0, stream>>>(agg_s, nS);
  fill_zero_kernel<<<2048, 256, 0, stream>>>(agg_d, nD);
  fill_zero_kernel<<<256, 256, 0, stream>>>(cnt_s, NSn);
  fill_zero_kernel<<<256, 256, 0, stream>>>(cnt_d, NDn);
  scatter_kernel<<<8192, 256, 0, stream>>>(xd_bf, xs_bf, esrc, edst,
                                           agg_s, agg_d, cnt_s, cnt_d, E, 1);
  mean_cvt_kernel<<<4096, 256, 0, stream>>>(agg_s, cnt_s, aggs_bf, NSn);
  mean_cvt_kernel<<<4096, 256, 0, stream>>>(agg_d, cnt_d, aggd_bf, NDn);
  sage_wmma_kernel<<<(NSn + 127) / 128, 256, 0, stream>>>(aggs_bf, xs_bf, wl1ds_t, wr1ds_t,
                                                          bl1_ds, z1s, NSn, 1);
  sage_wmma_kernel<<<(NDn + 127) / 128, 256, 0, stream>>>(aggd_bf, xd_bf, wl1sd_t, wr1sd_t,
                                                          bl1_sd, z1d, NDn, 1);

  // ---- layer 2 (counts reused) ----
  fill_zero_kernel<<<2048, 256, 0, stream>>>(agg_s, nS);
  fill_zero_kernel<<<2048, 256, 0, stream>>>(agg_d, nD);
  scatter_kernel<<<8192, 256, 0, stream>>>(z1d, z1s, esrc, edst,
                                           agg_s, agg_d, cnt_s, cnt_d, E, 0);
  mean_cvt_kernel<<<4096, 256, 0, stream>>>(agg_s, cnt_s, aggs_bf, NSn);
  mean_cvt_kernel<<<4096, 256, 0, stream>>>(agg_d, cnt_d, aggd_bf, NDn);
  sage_wmma_kernel<<<(NSn + 127) / 128, 256, 0, stream>>>(aggs_bf, z1s, wl2ds_t, wr2ds_t,
                                                          bl2_ds, z2s, NSn, 0);
  sage_wmma_kernel<<<(NDn + 127) / 128, 256, 0, stream>>>(aggd_bf, z1d, wl2sd_t, wr2sd_t,
                                                          bl2_sd, z2d, NDn, 0);

  // ---- fused edge decoder ----
  decoder_kernel<<<(EL + 63) / 64, 128, 0, stream>>>(z2d, z2s, lrow, lcol,
                                                     wdec1_t, bdec1, Wdec2, bdec2,
                                                     out, EL);
}
